// LocalLossBlockConv_52931176956337
// MI455X (gfx1250) — matrix operand order repaired and checked by reading, not compiled, not run on hardware
//
#include <hip/hip_runtime.h>

typedef __attribute__((ext_vector_type(16))) _Float16 v16h;
typedef __attribute__((ext_vector_type(8)))  _Float16 v8h;
typedef __attribute__((ext_vector_type(8)))  float    v8f;
typedef __attribute__((ext_vector_type(4)))  float    f4;   // true vector type for NT ld/st

#define THRESH 0.5f
#define DECAY  0.2f

// Problem configuration (fixed by setup_inputs)
constexpr int B_    = 128;
constexpr int CIN   = 3;
constexpr int COUT  = 128;
constexpr int HW    = 32;
constexpr int NCLS  = 10;
constexpr int TW    = 10;
constexpr int NPIX  = HW * HW;          // 1024
constexpr int NFEAT = COUT * NPIX;      // 131072 (decoder K dimension)
constexpr long NTOT = (long)B_ * NFEAT; // 16777216 (one spike slab)
constexpr int KCONV = CIN * 3 * 3;      // 27 (padded to 32 -> one WMMA)
constexpr int KSPLIT = 256;             // decoder K-chunks
constexpr int KCH    = NFEAT / KSPLIT;  // 512 columns per wave

// ---------------------------------------------------------------------------
// Conv 3x3 pad1 as implicit GEMM: D[16x16] = Wtile[16x27] x Im2col[27x16]
// One v_wmma_f32_16x16x32_f16 per tile.
// ---------------------------------------------------------------------------
__global__ void conv_wmma(const float* __restrict__ x,
                          const float* __restrict__ w,
                          float* __restrict__ h) {
  const int lane = threadIdx.x & 31;
  const int wave = threadIdx.x >> 5;
  const int tile = blockIdx.x * 8 + wave;     // 8192 blocks * 8 waves = 65536 tiles
  const int mt = tile & 7;                    // COUT tile
  const int nt = tile >> 3;                   // spatial tile
  const int m    = lane & 15;
  const int half = lane >> 4;
  const int co0 = mt * 16;
  const int n0  = nt * 16;

  // A: weights, row co0+m, K(i) = 16*half + i
  v16h a;
  const int co = co0 + m;
  #pragma unroll
  for (int i = 0; i < 16; ++i) {
    const int k = 16 * half + i;
    a[i] = (k < KCONV) ? (_Float16)w[co * KCONV + k] : (_Float16)0.0f;
  }

  // B: im2col column n = n0+m, K(i) = 16*half + i
  const int n  = n0 + m;
  const int b  = n >> 10;
  const int s  = n & 1023;
  const int y  = s >> 5;
  const int xx = s & 31;
  v16h bb;
  #pragma unroll
  for (int i = 0; i < 16; ++i) {
    const int k = 16 * half + i;
    _Float16 v = (_Float16)0.0f;
    if (k < KCONV) {
      const int ci = k / 9;
      const int r  = k - ci * 9;
      const int ky = r / 3;
      const int kx = r - ky * 3;
      const int yi = y + ky - 1;
      const int xi = xx + kx - 1;
      if (yi >= 0 && yi < HW && xi >= 0 && xi < HW)
        v = (_Float16)x[(b * CIN + ci) * NPIX + yi * HW + xi];
    }
    bb[i] = v;
  }

  v8f c = {};
  c = __builtin_amdgcn_wmma_f32_16x16x32_f16(false, a, false, bb,
                                             (short)0, c, false, false);
  #pragma unroll
  for (int r = 0; r < 8; ++r) {
    const int coo = co0 + r + 8 * half;
    h[(long)b * NFEAT + (long)coo * NPIX + s] = c[r];
  }
}

// ---------------------------------------------------------------------------
// Pre-convert decoder weights to f16, padded to 16 rows (rows >= NCLS zeroed)
// so the fused hot loop needs no masking and no f32->f16 conversions.
// dwh layout: [16][NFEAT]
// ---------------------------------------------------------------------------
__global__ void cvt_decw(const float* __restrict__ dw, _Float16* __restrict__ dwh) {
  const long i = (long)blockIdx.x * blockDim.x + threadIdx.x;
  if (i >= (long)16 * NFEAT) return;
  const int row = (int)(i / NFEAT);
  dwh[i] = (row < NCLS) ? (_Float16)dw[i] : (_Float16)0.0f;
}

// ---------------------------------------------------------------------------
// FUSED LIF + decoder GEMM.
// One wave per block owns batch-tile mt (16 rows) x K-chunk kc (512 cols).
// Per 32-wide K block: load h once (NT loads: read-once data), run 10-step
// LIF in registers, build the f16 A-matrix (spikes, exact in f16) directly,
// keep 10 accumulator sets so the B-tile is reused across all timesteps, and
// store f32 spikes with NON-TEMPORAL b128 stores (output is never re-read,
// keep 671 MB of dead writes out of the 192 MB L2).
// Deterministic K-partials (no atomics). partial layout: [t][kc][b][16]
// ---------------------------------------------------------------------------
__global__ void fused_lif_dec(const float* __restrict__ h,
                              const _Float16* __restrict__ dwh,
                              float* __restrict__ spikes,
                              float* __restrict__ partial) {
  const int lane = threadIdx.x;
  const int m    = lane & 15;       // A row within tile / D column (class)
  const int half = lane >> 4;
  const int kc = blockIdx.x;        // 0..KSPLIT-1
  const int mt = blockIdx.y;        // 0..7

  const int b = mt * 16 + m;
  const float*    __restrict__ hrow = h   + (long)b * NFEAT;
  const _Float16* __restrict__ wrow = dwh + (long)m * NFEAT;   // padded rows

  v8f c[TW];
  #pragma unroll
  for (int t = 0; t < TW; ++t) { v8f z = {}; c[t] = z; }

  const int k0 = kc * KCH;
  for (int kk = k0; kk < k0 + KCH; kk += 32) {
    // h for this lane's 16 elements (read exactly once -> non-temporal loads)
    const f4 f0 = __builtin_nontemporal_load((const f4*)(hrow + kk + 8 * half));
    const f4 f1 = __builtin_nontemporal_load((const f4*)(hrow + kk + 8 * half + 4));
    const f4 f2 = __builtin_nontemporal_load((const f4*)(hrow + kk + 16 + 8 * half));
    const f4 f3 = __builtin_nontemporal_load((const f4*)(hrow + kk + 16 + 8 * half + 4));
    float hv[16] = {f0.x, f0.y, f0.z, f0.w, f1.x, f1.y, f1.z, f1.w,
                    f2.x, f2.y, f2.z, f2.w, f3.x, f3.y, f3.z, f3.w};

    // B tile: K(i) = 8*half + i (i<8), 16 + 8*half + (i-8) (i>=8) -> two b128 loads
    const v8h blo = *(const v8h*)(wrow + kk + 8 * half);
    const v8h bhi = *(const v8h*)(wrow + kk + 16 + 8 * half);
    const v16h bb = __builtin_shufflevector(blo, bhi,
        0, 1, 2, 3, 4, 5, 6, 7, 8, 9, 10, 11, 12, 13, 14, 15);

    // prefetch next K block of h (global_prefetch_b8)
    __builtin_prefetch(hrow + kk + 32, 0, 3);

    float mem[16], spk[16];
    #pragma unroll
    for (int i = 0; i < 16; ++i) { mem[i] = 0.0f; spk[i] = 0.0f; }

    #pragma unroll
    for (int t = 0; t < TW; ++t) {
      v16h a;
      #pragma unroll
      for (int i = 0; i < 16; ++i) {
        mem[i] = mem[i] * DECAY + hv[i] - spk[i] * (THRESH * DECAY);
        spk[i] = (mem[i] > THRESH) ? 1.0f : 0.0f;
        a[i]   = (_Float16)spk[i];
      }
      // f32 spike output: write-once, never re-read -> non-temporal b128 stores
      float* so = spikes + (long)t * NTOT + (long)b * NFEAT + kk;
      f4 s0 = {spk[0],  spk[1],  spk[2],  spk[3]};
      f4 s1 = {spk[4],  spk[5],  spk[6],  spk[7]};
      f4 s2 = {spk[8],  spk[9],  spk[10], spk[11]};
      f4 s3 = {spk[12], spk[13], spk[14], spk[15]};
      __builtin_nontemporal_store(s0, (f4*)(so + 8 * half));
      __builtin_nontemporal_store(s1, (f4*)(so + 8 * half + 4));
      __builtin_nontemporal_store(s2, (f4*)(so + 16 + 8 * half));
      __builtin_nontemporal_store(s3, (f4*)(so + 16 + 8 * half + 4));

      c[t] = __builtin_amdgcn_wmma_f32_16x16x32_f16(false, a, false, bb,
                                                    (short)0, c[t], false, false);
    }
  }

  // D: row M = r + 8*half -> batch, col N = m -> class
  #pragma unroll
  for (int t = 0; t < TW; ++t) {
    #pragma unroll
    for (int r = 0; r < 8; ++r) {
      if (m < NCLS) {
        const int bout = mt * 16 + r + 8 * half;
        partial[(((long)t * KSPLIT + kc) * B_ + bout) * 16 + m] = c[t][r];
      }
    }
  }
}

// ---------------------------------------------------------------------------
// Deterministic parallel reduction of K-partials: dec[t][b][16]
// ---------------------------------------------------------------------------
__global__ void reduce_partial(const float* __restrict__ partial,
                               float* __restrict__ dec) {
  const int i = blockIdx.x * blockDim.x + threadIdx.x;   // over TW*B_*16
  if (i >= TW * B_ * 16) return;
  const int t   = i / (B_ * 16);
  const int rem = i - t * (B_ * 16);
  float s = 0.0f;
  for (int kc = 0; kc < KSPLIT; ++kc)
    s += partial[((long)t * KSPLIT + kc) * (B_ * 16) + rem];
  dec[i] = s;
}

// ---------------------------------------------------------------------------
// Output-layer LIF + MSE loss. One block, 128 threads (one per batch row).
// ---------------------------------------------------------------------------
__global__ void decode_loss(const float* __restrict__ dec,
                            const float* __restrict__ y_onehot,
                            float* __restrict__ out_loss) {
  __shared__ float red[128];
  const int b = threadIdx.x;
  float yhm[NCLS], yhs[NCLS];
  #pragma unroll
  for (int c = 0; c < NCLS; ++c) { yhm[c] = 0.0f; yhs[c] = 0.0f; }

  float acc = 0.0f;
  for (int t = 0; t < TW; ++t) {
    #pragma unroll
    for (int c = 0; c < NCLS; ++c) {
      const float dv = dec[((long)t * B_ + b) * 16 + c];
      yhm[c] = yhm[c] * DECAY + dv - yhs[c] * (THRESH * DECAY);
      yhs[c] = (yhm[c] > THRESH) ? 1.0f : 0.0f;
      const float d = yhs[c] - y_onehot[b * NCLS + c];
      acc += d * d;
    }
  }
  red[b] = acc;
  __syncthreads();
  for (int s = 64; s > 0; s >>= 1) {
    if (b < s) red[b] += red[b + s];
    __syncthreads();
  }
  if (b == 0) *out_loss = red[0] / (float)(B_ * NCLS);
}

// ---------------------------------------------------------------------------
extern "C" void kernel_launch(void* const* d_in, const int* in_sizes, int n_in,
                              void* d_out, int out_size, void* d_ws, size_t ws_size,
                              hipStream_t stream) {
  (void)in_sizes; (void)n_in; (void)out_size; (void)ws_size;
  const float* x        = (const float*)d_in[0];  // [128,3,32,32]
  const float* y_onehot = (const float*)d_in[2];  // [128,10]
  const float* conv_w   = (const float*)d_in[3];  // [128,3,3,3]
  const float* dec_w    = (const float*)d_in[4];  // [10,131072]

  float* out = (float*)d_out;                     // spikes [10*16777216] ++ loss

  // workspace layout (floats): h | dwh(f16, as 1Mi floats) | partial | dec
  float*    h       = (float*)d_ws;                       // NTOT floats (67 MB)
  _Float16* dwh     = (_Float16*)(h + NTOT);              // 16*NFEAT halves (4.2 MB)
  float*    partial = h + NTOT + (16 * NFEAT) / 2;        // TW*KSPLIT*B_*16 (21 MB)
  float*    dec     = partial + (long)TW * KSPLIT * B_ * 16; // TW*B_*16

  // 1) decoder weights -> f16, zero-padded to 16 rows
  cvt_decw<<<(16 * NFEAT + 255) / 256, 256, 0, stream>>>(dec_w, dwh);
  // 2) conv as implicit WMMA GEMM
  conv_wmma<<<8192, 256, 0, stream>>>(x, conv_w, h);
  // 3) fused LIF + decoder GEMM (spikes written once, never re-read)
  dim3 g(KSPLIT, 8);
  fused_lif_dec<<<g, 32, 0, stream>>>(h, dwh, out, partial);
  // 4) reduce K-partials
  reduce_partial<<<(TW * B_ * 16 + 255) / 256, 256, 0, stream>>>(partial, dec);
  // 5) output-layer LIF + MSE loss
  decode_loss<<<1, 128, 0, stream>>>(dec, y_onehot, out + (long)TW * NTOT);
}
